// stackedIndRNN_encoder_86517821212560
// MI455X (gfx1250) — matrix operand order repaired and checked by reading, not compiled, not run on hardware
//
#include <hip/hip_runtime.h>

typedef __attribute__((ext_vector_type(16))) __bf16 v16bf;
typedef __attribute__((ext_vector_type(8)))  float  v8f;

constexpr int kT = 2048, kB = 64, kI = 128, kH = 256, kL = 6, kC = 100;
constexpr int kM = kT * kB;                       // 131072 rows (T*B)
constexpr int kStatBlocks = 256;
constexpr int kRowsPerStatBlock = kM / kStatBlocks; // 512

// ---------- bf16 split helpers (bit-exact, no reliance on __bf16 arithmetic) ----------
__device__ __forceinline__ __bf16 f2bf(float f) {
  union { float f; unsigned u; } a; a.f = f;
  unsigned r = a.u + 0x7FFFu + ((a.u >> 16) & 1u);   // round-to-nearest-even
  unsigned short s = (unsigned short)(r >> 16);
  __bf16 b; __builtin_memcpy(&b, &s, 2);
  return b;
}
__device__ __forceinline__ float bf2f(__bf16 b) {
  unsigned short s; __builtin_memcpy(&s, &b, 2);
  union { unsigned u; float f; } a; a.u = ((unsigned)s) << 16;
  return a.f;
}

// ---------- operand swizzle kernels (run once per launch) ----------
// A-fragment layout (16x32 bf16 per chunk): d = ((atile*CH + kc)*32 + lane)*16 + e
//   row M = atile*16 + (lane&15);  K = kc*32 + [16*(v/4) + 8*(lane>>4) + 2*(v&3) + p], v=e/2,p=e&1
__global__ void conv_fragA(const float* __restrict__ src, __bf16* __restrict__ dhi,
                           __bf16* __restrict__ dlo, int K, int total) {
  int d = blockIdx.x * blockDim.x + threadIdx.x;
  if (d >= total) return;
  int e    = d & 15;
  int lane = (d >> 4) & 31;
  int rem  = d >> 9;
  int CH   = K >> 5;                 // chunks (power of two: 4 or 8)
  int kc   = rem & (CH - 1);
  int at   = rem / CH;
  int rr   = lane & 15, hf = lane >> 4;
  int v    = e >> 1, p = e & 1;
  int kp   = ((v >> 2) << 4) + (hf << 3) + ((v & 3) << 1) + p;
  int k    = (kc << 5) + kp;
  float x  = src[(size_t)(at * 16 + rr) * K + k];
  __bf16 bh = f2bf(x);
  dhi[d] = bh;
  dlo[d] = f2bf(x - bf2f(bh));
}

// B-fragment layout (32x16 bf16 per chunk): d = ((ntile*CH + kc)*32 + lane)*16 + e
//   N = ntile*16 + (lane&15);  K = kc*32 + 16*(e>>3) + 8*(lane>>4) + (e&7)
__global__ void conv_fragB(const float* __restrict__ src, __bf16* __restrict__ dhi,
                           __bf16* __restrict__ dlo, int K, int total) {
  int d = blockIdx.x * blockDim.x + threadIdx.x;
  if (d >= total) return;
  int e    = d & 15;
  int lane = (d >> 4) & 31;
  int rem  = d >> 9;
  int CH   = K >> 5;
  int kc   = rem & (CH - 1);
  int nt   = rem / CH;
  int n    = (nt << 4) + (lane & 15);
  int k    = (kc << 5) + ((e >> 3) << 4) + ((lane >> 4) << 3) + (e & 7);
  float x  = src[(size_t)k * kH + n];
  __bf16 bh = f2bf(x);
  dhi[d] = bh;
  dlo[d] = f2bf(x - bf2f(bh));
}

// ---------- WMMA GEMM: Y[M x 256] = A[M x K] * W[K x 256] + bias ----------
// Block = 8 waves; block owns one 16-row A tile and the full N=256 (2 n-tiles/wave).
// Inner loop is pure 32B vector loads + 6 wmma (3-term bf16 emulation, 2 n-tiles).
template <int CHUNKS>
__global__ void __launch_bounds__(256)
gemm_wmma(const __bf16* __restrict__ Ahi, const __bf16* __restrict__ Alo,
          const __bf16* __restrict__ Bhi, const __bf16* __restrict__ Blo,
          const float* __restrict__ bias, float* __restrict__ Y) {
  const int tid  = threadIdx.x;
  const int wave = tid >> 5;
  const int lane = tid & 31;
  const int hf   = lane >> 4;
  const int l16  = lane & 15;
  const int at   = blockIdx.x;      // A tile (16 rows)
  const int t0   = wave * 2;        // first of two n-tiles

  v8f acc0 = {}, acc1 = {};
  size_t aOff  = ((size_t)at * CHUNKS * 32 + lane) * 16;
  size_t b0Off = ((size_t)t0 * CHUNKS * 32 + lane) * 16;
  size_t b1Off = b0Off + (size_t)CHUNKS * 32 * 16;
#pragma unroll
  for (int kc = 0; kc < CHUNKS; ++kc) {
    v16bf ah  = *(const v16bf*)(Ahi + aOff);
    v16bf al  = *(const v16bf*)(Alo + aOff);
    v16bf bh0 = *(const v16bf*)(Bhi + b0Off);
    v16bf bl0 = *(const v16bf*)(Blo + b0Off);
    v16bf bh1 = *(const v16bf*)(Bhi + b1Off);
    v16bf bl1 = *(const v16bf*)(Blo + b1Off);
    acc0 = __builtin_amdgcn_wmma_f32_16x16x32_bf16(false, ah, false, bh0, (short)0, acc0, false, false);
    acc1 = __builtin_amdgcn_wmma_f32_16x16x32_bf16(false, ah, false, bh1, (short)0, acc1, false, false);
    acc0 = __builtin_amdgcn_wmma_f32_16x16x32_bf16(false, ah, false, bl0, (short)0, acc0, false, false);
    acc1 = __builtin_amdgcn_wmma_f32_16x16x32_bf16(false, ah, false, bl1, (short)0, acc1, false, false);
    acc0 = __builtin_amdgcn_wmma_f32_16x16x32_bf16(false, al, false, bh0, (short)0, acc0, false, false);
    acc1 = __builtin_amdgcn_wmma_f32_16x16x32_bf16(false, al, false, bh1, (short)0, acc1, false, false);
    aOff += 512; b0Off += 512; b1Off += 512;
  }
  const int c0 = t0 * 16 + l16;
  const int c1 = c0 + 16;
  const float bb0 = bias[c0], bb1 = bias[c1];
#pragma unroll
  for (int v = 0; v < 8; ++v) {     // D: VGPR v -> row v + 8*hf, col = lane&15
    int row = at * 16 + v + hf * 8;
    Y[(size_t)row * kH + c0] = acc0[v] + bb0;
    Y[(size_t)row * kH + c1] = acc1[v] + bb1;
  }
}

// ---------- BN stats: deterministic two-pass ----------
__global__ void __launch_bounds__(256)
bn_stats_part(const float* __restrict__ Y, float* __restrict__ psum, float* __restrict__ psq) {
  int c = threadIdx.x;
  size_t r0 = (size_t)blockIdx.x * kRowsPerStatBlock;
  float s = 0.f, q = 0.f;
  for (int i = 0; i < kRowsPerStatBlock; ++i) {
    float v = Y[(r0 + i) * kH + c];
    s += v;
    q = fmaf(v, v, q);
  }
  psum[blockIdx.x * kH + c] = s;
  psq [blockIdx.x * kH + c] = q;
}

__global__ void bn_finalize(const float* __restrict__ psum, const float* __restrict__ psq,
                            const float* __restrict__ gamma, const float* __restrict__ beta,
                            float* __restrict__ scale, float* __restrict__ shift) {
  int c = threadIdx.x;
  float s = 0.f, q = 0.f;
  for (int j = 0; j < kStatBlocks; ++j) { s += psum[j * kH + c]; q += psq[j * kH + c]; }
  float mean = s * (1.0f / kM);
  float var  = q * (1.0f / kM) - mean * mean;   // biased, matches reference
  float rstd = rsqrtf(var + 1e-5f);
  float sc   = gamma[c] * rstd;
  scale[c] = sc;
  shift[c] = fmaf(-mean, sc, beta[c]);
}

// ---------- fused BN-apply + IndRNN scan; emits next layer's A fragments ----------
__global__ void __launch_bounds__(256)
indrnn_scan(const float* __restrict__ Y, const float* __restrict__ scale,
            const float* __restrict__ shift, const float* __restrict__ u,
            __bf16* __restrict__ Ahi, __bf16* __restrict__ Alo,
            float* __restrict__ hlast, int writeNext) {
  const int g = blockIdx.x * 256 + threadIdx.x;  // (b, c), 16384 lanes
  const int b = g >> 8;
  const int c = g & 255;
  const float sc = scale[c], sh = shift[c], uu = u[c];
  // precompute swizzled A-fragment slot for (row = t*64 + b, channel c), K=256 -> CH=8
  const int rr    = b & 15;
  const int btile = b >> 4;                      // Atile = t*4 + btile
  const int kc    = c >> 5;
  const int kp    = c & 31;
  const int hfa   = (kp >> 3) & 1;
  const int v     = ((kp >> 4) << 2) + ((kp & 7) >> 1);
  const int e     = (v << 1) + (kp & 1);
  const int lanea = (hfa << 4) + rr;
  size_t dOff = (((size_t)btile * 8 + kc) * 32 + lanea) * 16 + e;
  size_t sIdx = (size_t)g;
  float h = 0.f;
#pragma unroll 4
  for (int t = 0; t < kT; ++t) {
    float y = fmaf(Y[sIdx], sc, sh);
    h = fmaxf(fmaf(uu, h, y), 0.f);
    if (writeNext) {
      __bf16 bh = f2bf(h);
      Ahi[dOff] = bh;
      Alo[dOff] = f2bf(h - bf2f(bh));
    }
    sIdx += (size_t)kB * kH;                     // +16384 per timestep
    dOff += (size_t)4 * 8 * 512;                 // 4 A-tiles per timestep
  }
  if (hlast) hlast[g] = h;                       // h at t = T-1
}

// ---------- tiny classifier: out[64x100] = hlast[64x256] @ Wc + bc ----------
__global__ void classifier(const float* __restrict__ Hl, const float* __restrict__ Wc,
                           const float* __restrict__ bc, float* __restrict__ out) {
  int i = blockIdx.x * blockDim.x + threadIdx.x;
  if (i >= kB * kC) return;
  int b = i / kC, c = i - b * kC;
  float s = bc[c];
  for (int k = 0; k < kH; ++k) s = fmaf(Hl[b * kH + k], Wc[k * kC + c], s);
  out[i] = s;
}

extern "C" void kernel_launch(void* const* d_in, const int* in_sizes, int n_in,
                              void* d_out, int out_size, void* d_ws, size_t ws_size,
                              hipStream_t stream) {
  (void)in_sizes; (void)n_in; (void)out_size; (void)ws_size;
  const float* x  = (const float*)d_in[0];
  const float* W0 = (const float*)d_in[1];
  const float* b0 = (const float*)d_in[2];
  const float* Wl = (const float*)d_in[3];
  const float* bl = (const float*)d_in[4];
  const float* u  = (const float*)d_in[5];
  const float* gm = (const float*)d_in[6];
  const float* bt = (const float*)d_in[7];
  const float* Wc = (const float*)d_in[8];
  const float* bc = (const float*)d_in[9];
  float* out = (float*)d_out;

  // ---- workspace layout (bytes) ----
  char* ws = (char*)d_ws;
  float*  Y     = (float*) (ws);                       // 131072*256*4 = 134217728
  __bf16* Ahi   = (__bf16*)(ws + 134217728ull);        // 131072*256*2 =  67108864
  __bf16* Alo   = (__bf16*)(ws + 201326592ull);        //                 67108864
  __bf16* Bhi   = (__bf16*)(ws + 268435456ull);        // 360448*2     =    720896
  __bf16* Blo   = (__bf16*)(ws + 269156352ull);        //                   720896
  float*  psum  = (float*) (ws + 269877248ull);        // 256*256*4    =    262144
  float*  psq   = (float*) (ws + 270139392ull);        //                   262144
  float*  scl   = (float*) (ws + 270401536ull);        // 256*4
  float*  shf   = (float*) (ws + 270402560ull);        // 256*4
  float*  hlast = (float*) (ws + 270403584ull);        // 64*256*4

  // ---- one-time per-launch operand swizzles ----
  conv_fragB<<<(kI * kH) / 256, 256, 0, stream>>>(W0, Bhi, Blo, kI, kI * kH);
  for (int l = 1; l < kL; ++l) {
    size_t off = (size_t)kI * kH + (size_t)(l - 1) * kH * kH;
    conv_fragB<<<(kH * kH) / 256, 256, 0, stream>>>(Wl + (size_t)(l - 1) * kH * kH,
                                                    Bhi + off, Blo + off, kH, kH * kH);
  }
  conv_fragA<<<(kM * kI) / 256, 256, 0, stream>>>(x, Ahi, Alo, kI, kM * kI);

  // ---- layer pipeline ----
  for (int l = 0; l < kL; ++l) {
    size_t wOff = (l == 0) ? 0 : (size_t)kI * kH + (size_t)(l - 1) * kH * kH;
    const float* bias = (l == 0) ? b0 : bl + (size_t)(l - 1) * kH;
    if (l == 0)
      gemm_wmma<kI / 32><<<kM / 16, 256, 0, stream>>>(Ahi, Alo, Bhi + wOff, Blo + wOff, bias, Y);
    else
      gemm_wmma<kH / 32><<<kM / 16, 256, 0, stream>>>(Ahi, Alo, Bhi + wOff, Blo + wOff, bias, Y);
    bn_stats_part<<<kStatBlocks, 256, 0, stream>>>(Y, psum, psq);
    bn_finalize<<<1, kH, 0, stream>>>(psum, psq, gm + l * kH, bt + l * kH, scl, shf);
    const int last = (l == kL - 1);
    indrnn_scan<<<(kB * kH) / 256, 256, 0, stream>>>(Y, scl, shf, u + l * kH,
                                                     Ahi, Alo, last ? hlast : nullptr,
                                                     last ? 0 : 1);
  }
  classifier<<<(kB * kC + 255) / 256, 256, 0, stream>>>(hlast, Wc, bc, out);
}
